// MambaBlock_87531433492526
// MI455X (gfx1250) — matrix operand order repaired and checked by reading, not compiled, not run on hardware
//
#include <hip/hip_runtime.h>

#define TT    2048
#define DM    768
#define DI    1536
#define NS    16
#define DIN   (DI * NS)   // 24576

typedef __attribute__((ext_vector_type(16))) __bf16 v16bf;
typedef __attribute__((ext_vector_type(8)))  float  v8f;

union FragBF { unsigned u[8]; v16bf v; };

__device__ __forceinline__ unsigned short f2bf(float x) {
    unsigned u = __float_as_uint(x);
    unsigned r = u + 0x7FFFu + ((u >> 16) & 1u);   // round-to-nearest-even
    return (unsigned short)(r >> 16);
}
__device__ __forceinline__ float bf2f(unsigned short h) {
    return __uint_as_float(((unsigned)h) << 16);
}

// ---------------------------------------------------------------- f32 -> bf16 bulk convert
__global__ __launch_bounds__(256) void cvt_bf16_kernel(
    const float* __restrict__ src, unsigned short* __restrict__ dst)
{
    const size_t i = ((size_t)blockIdx.x * 256 + threadIdx.x) * 8;
    const float4 f0 = *(const float4*)(src + i);
    const float4 f1 = *(const float4*)(src + i + 4);
    uint4 o;
    o.x = (unsigned)f2bf(f0.x) | ((unsigned)f2bf(f0.y) << 16);
    o.y = (unsigned)f2bf(f0.z) | ((unsigned)f2bf(f0.w) << 16);
    o.z = (unsigned)f2bf(f1.x) | ((unsigned)f2bf(f1.y) << 16);
    o.w = (unsigned)f2bf(f1.z) | ((unsigned)f2bf(f1.w) << 16);
    *(uint4*)(dst + i) = o;
}

// ---------------------------------------------------------------- rmsnorm -> bf16
__global__ __launch_bounds__(256) void rmsnorm_kernel(
    const float* __restrict__ xs, const float* __restrict__ w,
    unsigned short* __restrict__ xb)
{
    const int t = blockIdx.x;
    const float* row = xs + (size_t)t * DM;
    __shared__ float red[256];
    float ss = 0.0f;
    for (int c = threadIdx.x; c < DM; c += 256) { float v = row[c]; ss += v * v; }
    red[threadIdx.x] = ss;
    __syncthreads();
    for (int s = 128; s > 0; s >>= 1) {
        if (threadIdx.x < s) red[threadIdx.x] += red[threadIdx.x + s];
        __syncthreads();
    }
    const float inv = rsqrtf(red[0] / (float)DM + 1e-5f);
    for (int c = threadIdx.x; c < DM; c += 256)
        xb[(size_t)t * DM + c] = f2bf(row[c] * inv * w[c]);
}

// ---------------------------------------------------------------- causal depthwise conv
__global__ __launch_bounds__(256) void conv_kernel(
    const float* __restrict__ u, const float* __restrict__ cw,
    const float* __restrict__ cb, float* __restrict__ uc,
    unsigned short* __restrict__ ucb)
{
    const int idx = blockIdx.x * 256 + threadIdx.x;   // T*DI exact
    const int t = idx / DI;
    const int c = idx - t * DI;
    float acc = cb[c];
#pragma unroll
    for (int j = 0; j < 4; ++j) {
        const int tt = t - 3 + j;
        if (tt >= 0) acc += u[(size_t)tt * DI + c] * cw[c * 4 + j];
    }
    uc[idx]  = acc;
    ucb[idx] = f2bf(acc);
}

// ---------------------------------------------------------------- bf16 WMMA GEMM
// out = epilogue( A[M,K](bf16) @ W[N,K]^T(bf16) + bias )
// MODE 1: split -> out0=f32 (n<split), out1=sigmoid f32 (n>=split), both stride=split
// MODE 2: softplus -> out0 f32 stride N
// MODE 3: bf16 -> out0 stride N
// MODE 4: residual -> out0 f32 = v + aux, stride N
#define BM 128
#define BN 64
#define BK 32
#define LDT 34

__device__ __forceinline__ v8f wmma_bf16(const FragBF& a, const FragBF& b, v8f c) {
    return __builtin_amdgcn_wmma_f32_16x16x32_bf16(false, a.v, false, b.v,
                                                   (short)0, c, false, false);
}

template <int MODE>
__global__ __launch_bounds__(256) void gemm_bf16_kernel(
    const unsigned short* __restrict__ A, const unsigned short* __restrict__ W,
    const float* __restrict__ bias, void* __restrict__ out0,
    void* __restrict__ out1, const float* __restrict__ aux,
    int M, int N, int K, int split)
{
    __shared__ unsigned short As[2][BM * LDT];
    __shared__ unsigned short Ws[2][BN * LDT];

    const int tid  = threadIdx.x;
    const int lane = tid & 31;
    const int wid  = tid >> 5;
    const int half = lane >> 4;
    const int l15  = lane & 15;
    const int m0 = blockIdx.y * BM;
    const int n0 = blockIdx.x * BN;
    const int mw = (wid & 3) * 32;     // wave's 32-row sub-block
    const int nw = (wid >> 2) * 32;    // wave's 32-col sub-block

    // staging mappings (pure 16B copies, A: 2 per thread, W: 1 per thread)
    const int ar = tid >> 1;           // 0..127 (A tile row)
    const int ac = (tid & 1) * 16;     // A col segment (16 bf16 = 32B)
    const int wr = tid >> 2;           // 0..63  (W tile row)
    const int wc = (tid & 3) * 8;      // W col segment (8 bf16 = 16B)

    const unsigned short* aptr = A + (size_t)(m0 + ar) * K + ac;
    const unsigned short* wptr = W + (size_t)(n0 + wr) * K + wc;

    v8f acc[2][2] = {};

    // prologue: stage k0 = 0 into buffer 0
    {
        uint4 av0 = *(const uint4*)(aptr);
        uint4 av1 = *(const uint4*)(aptr + 8);
        uint4 wv  = *(const uint4*)(wptr);
        unsigned* d = (unsigned*)&As[0][ar * LDT + ac];
        d[0] = av0.x; d[1] = av0.y; d[2] = av0.z; d[3] = av0.w;
        d[4] = av1.x; d[5] = av1.y; d[6] = av1.z; d[7] = av1.w;
        unsigned* dw = (unsigned*)&Ws[0][wr * LDT + wc];
        dw[0] = wv.x; dw[1] = wv.y; dw[2] = wv.z; dw[3] = wv.w;
    }
    __syncthreads();

    int buf = 0;
    for (int k0 = 0; k0 < K; k0 += BK) {
        const bool has_next = (k0 + BK) < K;
        uint4 av0, av1, wv;
        if (has_next) {
            const unsigned short* ap = aptr + k0 + BK;
            av0 = *(const uint4*)ap;
            av1 = *(const uint4*)(ap + 8);
            wv  = *(const uint4*)(wptr + k0 + BK);
            if (k0 + 2 * BK < K) {
                __builtin_prefetch(aptr + k0 + 2 * BK, 0, 1);
                __builtin_prefetch(wptr + k0 + 2 * BK, 0, 1);
            }
        }

        // fragments from LDS[buf]
        FragBF a0, a1, b0, b1;
#pragma unroll
        for (int j = 0; j < 8; ++j) {
            const int kk = (j < 4) ? (half * 8 + 2 * j)
                                   : (16 + half * 8 + 2 * (j - 4));
            a0.u[j] = *(const unsigned*)&As[buf][(mw + l15) * LDT + kk];
            a1.u[j] = *(const unsigned*)&As[buf][(mw + 16 + l15) * LDT + kk];
        }
#pragma unroll
        for (int j = 0; j < 8; ++j) {
            const int kk = (j < 4) ? (half * 8 + 2 * j)
                                   : (16 + half * 8 + 2 * (j - 4));
            b0.u[j] = *(const unsigned*)&Ws[buf][(nw + l15) * LDT + kk];
            b1.u[j] = *(const unsigned*)&Ws[buf][(nw + 16 + l15) * LDT + kk];
        }
        acc[0][0] = wmma_bf16(a0, b0, acc[0][0]);
        acc[0][1] = wmma_bf16(a0, b1, acc[0][1]);
        acc[1][0] = wmma_bf16(a1, b0, acc[1][0]);
        acc[1][1] = wmma_bf16(a1, b1, acc[1][1]);

        if (has_next) {
            unsigned* d = (unsigned*)&As[buf ^ 1][ar * LDT + ac];
            d[0] = av0.x; d[1] = av0.y; d[2] = av0.z; d[3] = av0.w;
            d[4] = av1.x; d[5] = av1.y; d[6] = av1.z; d[7] = av1.w;
            unsigned* dw = (unsigned*)&Ws[buf ^ 1][wr * LDT + wc];
            dw[0] = wv.x; dw[1] = wv.y; dw[2] = wv.z; dw[3] = wv.w;
        }
        __syncthreads();
        buf ^= 1;
    }

    // epilogue
#pragma unroll
    for (int mt = 0; mt < 2; ++mt) {
#pragma unroll
        for (int nt = 0; nt < 2; ++nt) {
            const v8f a = acc[mt][nt];
            const int n_g = n0 + nw + nt * 16 + l15;
            const float bv = bias[n_g];
#pragma unroll
            for (int r = 0; r < 8; ++r) {
                const int m_g = m0 + mw + mt * 16 + 8 * half + r;
                const float v = a[r] + bv;
                if constexpr (MODE == 1) {
                    if (n_g < split)
                        ((float*)out0)[(size_t)m_g * split + n_g] = v;
                    else
                        ((float*)out1)[(size_t)m_g * split + (n_g - split)] =
                            1.0f / (1.0f + __expf(-v));
                } else if constexpr (MODE == 2) {
                    ((float*)out0)[(size_t)m_g * N + n_g] =
                        (v > 20.0f) ? v : log1pf(__expf(v));
                } else if constexpr (MODE == 3) {
                    ((unsigned short*)out0)[(size_t)m_g * N + n_g] = f2bf(v);
                } else if constexpr (MODE == 4) {
                    ((float*)out0)[(size_t)m_g * N + n_g] =
                        v + aux[(size_t)m_g * N + n_g];
                } else {
                    ((float*)out0)[(size_t)m_g * N + n_g] = v;
                }
            }
        }
    }
}

// ---------------------------------------------------------------- selective scan
// 16 lanes per channel, one SSM state per lane; shfl-xor reduction over states.
__global__ __launch_bounds__(256) void scan_kernel(
    const float* __restrict__ delta, const float* __restrict__ u,
    const unsigned short* __restrict__ Bm, const unsigned short* __restrict__ Cm,
    const float* __restrict__ A_log, const float* __restrict__ D_skip,
    float* __restrict__ y)
{
    const int lane = threadIdx.x & 31;
    const int wid  = threadIdx.x >> 5;
    const int n    = lane & 15;
    const int chan = blockIdx.x * 16 + wid * 2 + (lane >> 4);

    const float An = -log1pf(__expf(A_log[n]));   // A = -softplus(A_log)
    const float Dc = D_skip[chan];
    float state = 0.0f;

    for (int t = 0; t < TT; ++t) {
        const float d  = delta[(size_t)t * DI + chan];
        const float uu = u[(size_t)t * DI + chan];
        const float b  = bf2f(Bm[(size_t)t * DIN + chan * NS + n]);
        const float c  = bf2f(Cm[(size_t)t * DIN + chan * NS + n]);
        const float a  = __expf(d * An);
        state = fmaf(a, state, b * uu);
        float val = state * c;
        val += __shfl_xor(val, 8);
        val += __shfl_xor(val, 4);
        val += __shfl_xor(val, 2);
        val += __shfl_xor(val, 1);
        if (n == 0) y[(size_t)t * DI + chan] = val + Dc * uu;
    }
}

// ---------------------------------------------------------------- gate: h = sigmoid(z)*y -> bf16
__global__ __launch_bounds__(256) void gate_kernel(
    const float* __restrict__ s, const float* __restrict__ y,
    unsigned short* __restrict__ h)
{
    const int idx = blockIdx.x * 256 + threadIdx.x;
    h[idx] = f2bf(s[idx] * y[idx]);
}

// ---------------------------------------------------------------- launch
extern "C" void kernel_launch(void* const* d_in, const int* in_sizes, int n_in,
                              void* d_out, int out_size, void* d_ws, size_t ws_size,
                              hipStream_t stream)
{
    const float* xs      = (const float*)d_in[0];
    const float* norm_w  = (const float*)d_in[1];
    const float* in_w    = (const float*)d_in[2];
    const float* in_b    = (const float*)d_in[3];
    const float* conv_w  = (const float*)d_in[4];
    const float* conv_b  = (const float*)d_in[5];
    const float* delta_w = (const float*)d_in[6];
    const float* delta_b = (const float*)d_in[7];
    const float* B_w     = (const float*)d_in[8];
    const float* B_b     = (const float*)d_in[9];
    const float* C_w     = (const float*)d_in[10];
    const float* C_b     = (const float*)d_in[11];
    const float* A_log   = (const float*)d_in[12];
    const float* D_skip  = (const float*)d_in[13];
    const float* out_w   = (const float*)d_in[14];
    const float* out_b   = (const float*)d_in[15];

    char* ws = (char*)d_ws;
    size_t off = 0;
    auto alloc = [&](size_t bytes) -> char* {
        char* p = ws + off;
        off = (off + bytes + 255) & ~(size_t)255;
        return p;
    };
    unsigned short* x_bf = (unsigned short*)alloc((size_t)TT * DM * 2);
    float*          u_f  = (float*)         alloc((size_t)TT * DI * 4);
    float*          s_f  = (float*)         alloc((size_t)TT * DI * 4);
    float*          uc_f = (float*)         alloc((size_t)TT * DI * 4);
    unsigned short* ucb  = (unsigned short*)alloc((size_t)TT * DI * 2);
    float*          dl_f = (float*)         alloc((size_t)TT * DI * 4);
    unsigned short* Bm   = (unsigned short*)alloc((size_t)TT * DIN * 2);
    unsigned short* Cm   = (unsigned short*)alloc((size_t)TT * DIN * 2);
    float*          y_f  = (float*)         alloc((size_t)TT * DI * 4);
    unsigned short* h_bf = (unsigned short*)alloc((size_t)TT * DI * 2);
    // bf16 weight copies (converted once per launch)
    unsigned short* in_w_bf  = (unsigned short*)alloc((size_t)2 * DI * DM * 2);
    unsigned short* dl_w_bf  = (unsigned short*)alloc((size_t)DI * DI * 2);
    unsigned short* B_w_bf   = (unsigned short*)alloc((size_t)DIN * DI * 2);
    unsigned short* C_w_bf   = (unsigned short*)alloc((size_t)DIN * DI * 2);
    unsigned short* out_w_bf = (unsigned short*)alloc((size_t)DM * DI * 2);

    // 0) weight conversion passes (element counts all divisible by 2048)
    cvt_bf16_kernel<<<(2 * DI * DM) / 2048, 256, 0, stream>>>(in_w, in_w_bf);
    cvt_bf16_kernel<<<(DI * DI) / 2048, 256, 0, stream>>>(delta_w, dl_w_bf);
    cvt_bf16_kernel<<<(DIN * DI) / 2048, 256, 0, stream>>>(B_w, B_w_bf);
    cvt_bf16_kernel<<<(DIN * DI) / 2048, 256, 0, stream>>>(C_w, C_w_bf);
    cvt_bf16_kernel<<<(DM * DI) / 2048, 256, 0, stream>>>(out_w, out_w_bf);

    // 1) RMSNorm -> bf16
    rmsnorm_kernel<<<TT, 256, 0, stream>>>(xs, norm_w, x_bf);

    // 2) in_proj: [T,768] x [3072,768]^T -> u (f32) | sigmoid(z) (f32)
    gemm_bf16_kernel<1><<<dim3(2 * DI / BN, TT / BM), 256, 0, stream>>>(
        x_bf, in_w_bf, in_b, u_f, s_f, nullptr, TT, 2 * DI, DM, DI);

    // 3) causal depthwise conv -> uc (f32) + bf16 copy
    conv_kernel<<<(TT * DI) / 256, 256, 0, stream>>>(u_f, conv_w, conv_b, uc_f, ucb);

    // 4) delta = softplus(uc @ delta_w^T + b)
    gemm_bf16_kernel<2><<<dim3(DI / BN, TT / BM), 256, 0, stream>>>(
        ucb, dl_w_bf, delta_b, dl_f, nullptr, nullptr, TT, DI, DI, 0);

    // 5/6) B and C projections -> bf16 [T, 24576]
    gemm_bf16_kernel<3><<<dim3(DIN / BN, TT / BM), 256, 0, stream>>>(
        ucb, B_w_bf, B_b, Bm, nullptr, nullptr, TT, DIN, DI, 0);
    gemm_bf16_kernel<3><<<dim3(DIN / BN, TT / BM), 256, 0, stream>>>(
        ucb, C_w_bf, C_b, Cm, nullptr, nullptr, TT, DIN, DI, 0);

    // 7) selective scan (16 lanes per channel)
    scan_kernel<<<DI / 16, 256, 0, stream>>>(dl_f, uc_f, Bm, Cm, A_log, D_skip, y_f);

    // 8) gate h = sigmoid(z) * y -> bf16
    gate_kernel<<<(TT * DI) / 256, 256, 0, stream>>>(s_f, y_f, h_bf);

    // 9) out = xs + h @ out_w^T + out_b
    gemm_bf16_kernel<4><<<dim3(DM / BN, TT / BM), 256, 0, stream>>>(
        h_bf, out_w_bf, out_b, d_out, nullptr, xs, TT, DM, DI, 0);
}
